// DGCN_29755533426724
// MI455X (gfx1250) — compile-verified, matrix operand order checked by reading, not run on hardware
//
#include <hip/hip_runtime.h>
#include <hip/hip_bf16.h>

// ---------------- problem constants ----------------
#define B_    32
#define C_    128
#define N_    883
#define T_    12
#define NP    896          // N padded to multiple of 128
#define KTOP  706          // int(0.8*N)
#define SCALE 0.08838834764831845f   // 1/sqrt(128)

typedef __attribute__((ext_vector_type(16))) __bf16 bf16x16;
typedef __attribute__((ext_vector_type(8)))  float  floatx8;
typedef __attribute__((ext_vector_type(8)))  unsigned short ushort8;  // 16B
typedef int v4i_gcc __attribute__((__vector_size__(16)));             // int __vector(4)

// CDNA5 async global->LDS path (ASYNCcnt), with safe fallback
#if defined(__gfx1250__) && __has_builtin(__builtin_amdgcn_global_load_async_to_lds_b128)
#define ASYNC_LDS 1
#else
#define ASYNC_LDS 0
#endif

#if ASYNC_LDS
__device__ __forceinline__ void async_cp16(const unsigned short* g, unsigned short* l) {
  // as(1) ptr = 64-bit VA; as(3) ptr = LDS offset (low 32 bits of flat shared VA)
  auto gp = (__attribute__((address_space(1))) v4i_gcc*)(unsigned long long)(__SIZE_TYPE__)g;
  auto lp = (__attribute__((address_space(3))) v4i_gcc*)(unsigned)(__SIZE_TYPE__)l;
  __builtin_amdgcn_global_load_async_to_lds_b128(gp, lp, 0, 0);
}
#endif
__device__ __forceinline__ void wait_async0() {
#if defined(__gfx1250__)
#if __has_builtin(__builtin_amdgcn_s_wait_asynccnt)
  __builtin_amdgcn_s_wait_asynccnt(0);
#else
  asm volatile("s_wait_asynccnt 0" ::: "memory");
#endif
#endif
}

// ---------------- bf16 helpers (bit-exact RNE, storage = ushort) ------------
__device__ __forceinline__ unsigned short f2bfu(float f) {
  unsigned u = __builtin_bit_cast(unsigned, f);
  unsigned r = u + 0x7FFFu + ((u >> 16) & 1u);
  return (unsigned short)(r >> 16);
}
__device__ __forceinline__ float bfu2f(unsigned short h) {
  unsigned u = ((unsigned)h) << 16;
  return __builtin_bit_cast(float, u);
}

// ---------------- GEMM: C(f32 acc) = A(bf16) * B(bf16), batched ------------
// block = 128 threads = 4 waves; block tile 64(M) x 128(N) x 32(K);
// wave tile 32x64 = 2x4 x v_wmma_f32_16x16x32_bf16 per K-step.
// Register-staged software pipeline: tile k+1 is fetched while tile k computes.
// All dims are exact multiples of the tile (enforced by the launcher) -> no guards.
enum { EPI_H = 0, EPI_F32 = 1, EPI_G = 2, EPI_OUT = 3 };

struct GemmP {
  const unsigned short* A; int rsA; int csA; long long batchA; int divA;
  const unsigned short* B; int rsB;          long long batchB; int divB;
  int K;
  void* out; int ldO; long long batchO;
  const float* bias; const float* emb; const float* skip;
};

#define LDSP 40   // padded LDS row stride (bf16 elems) to spread banks

__device__ __forceinline__ bf16x16 ldfrag(const unsigned short* row, int half) {
  // ISA 7.12.2 16-bit A layout: lane l: M=l&15; elems 0..7 -> K = 8*(l>>4)+e,
  // elems 8..15 -> K = 16 + 8*(l>>4) + (e-8). B fragment symmetric (N=l&15).
  bf16x16 f;
  const unsigned short* p0 = row + half * 8;
  const unsigned short* p1 = row + 16 + half * 8;
#pragma unroll
  for (int e = 0; e < 8; ++e) {
    f[e]     = __builtin_bit_cast(__bf16, p0[e]);
    f[e + 8] = __builtin_bit_cast(__bf16, p1[e]);
  }
  return f;
}

template <int EPI, bool ACOL>
__global__ __launch_bounds__(128) void gemm_bf16_k(GemmP p) {
  __shared__ unsigned short As[64 * LDSP];    //  5120 B
  __shared__ unsigned short Bs[128 * LDSP];   // 10240 B

  const int tid  = threadIdx.x;
  const int lane = tid & 31;
  const int wave = tid >> 5;
  const int lm   = lane & 15;
  const int half = lane >> 4;
  const int wm   = (wave >> 1) * 32;
  const int wn   = (wave & 1) * 64;
  const int z    = blockIdx.z;
  const int i0   = blockIdx.y * 64;
  const int j0   = blockIdx.x * 128;

  // per-batch bases (64-bit, scalar)
  const unsigned short* __restrict__ Ap =
      p.A + (long long)(z / p.divA) * p.batchA + (ACOL ? (long long)i0 : (long long)i0 * p.rsA);
  const unsigned short* __restrict__ Bp =
      p.B + (long long)(z / p.divB) * p.batchB + j0;

  // ---- precomputed per-thread offsets (32-bit) ----
  // A fast path: 2048 elems = 128 thr x 2 x ushort8. chunk c: m=c>>2, kc=(c&3)*8
  const int mA  = tid >> 2;
  const int kA  = (tid & 3) * 8;
  const int offA0 = mA * p.rsA + kA;            // row-major only
  const int offA1 = offA0 + 32 * p.rsA;
  unsigned short* sA0 = &As[mA * LDSP + kA];
  unsigned short* sA1 = sA0 + 32 * LDSP;
  // B: 32k x 128n = 4096 elems = 128 thr x 4 x ushort8 (coalesced in n)
  const int kB  = tid >> 4;                     // 0..7
  const int nB  = (tid & 15) * 8;               // 0..120
  unsigned short* sB = &Bs[nB * LDSP + kB];     // transposed store base

  floatx8 acc[2][4];
#pragma unroll
  for (int a = 0; a < 2; ++a)
#pragma unroll
    for (int b = 0; b < 4; ++b)
#pragma unroll
      for (int e = 0; e < 8; ++e) acc[a][b][e] = 0.0f;

  // ---- pipeline prologue: stage tile 0 in registers ----
  ushort8 ra0, ra1, rb0, rb1, rb2, rb3;
  unsigned short rac[16];
  if (!ACOL) {
#if !ASYNC_LDS
    ra0 = *(const ushort8*)(Ap + offA0);
    ra1 = *(const ushort8*)(Ap + offA1);
    Ap += 32;
#endif
  } else {
#pragma unroll
    for (int r = 0; r < 16; ++r) {
      int el = tid + r * 128;
      rac[r] = Ap[(el & 63) + (el >> 6) * p.csA];
    }
    Ap += 32 * p.csA;
  }
  rb0 = *(const ushort8*)(Bp + (kB + 0)  * p.rsB + nB);
  rb1 = *(const ushort8*)(Bp + (kB + 8)  * p.rsB + nB);
  rb2 = *(const ushort8*)(Bp + (kB + 16) * p.rsB + nB);
  rb3 = *(const ushort8*)(Bp + (kB + 24) * p.rsB + nB);
  Bp += 32 * p.rsB;

  for (int kb = 0; kb < p.K; kb += 32) {
    // ---- publish tile kb to LDS ----
    if (!ACOL) {
#if ASYNC_LDS
      async_cp16(Ap + offA0, sA0);   // global -> LDS, tracked by ASYNCcnt
      async_cp16(Ap + offA1, sA1);
      Ap += 32;
#else
      *(ushort8*)sA0 = ra0;
      *(ushort8*)sA1 = ra1;
#endif
    } else {
#pragma unroll
      for (int r = 0; r < 16; ++r) {
        int el = tid + r * 128;
        As[(el & 63) * LDSP + (el >> 6)] = rac[r];
      }
    }
    {  // B transposed scatter Bs[n][k]
      unsigned short* s0 = sB;
#pragma unroll
      for (int e = 0; e < 8; ++e) s0[e * LDSP + 0]  = rb0[e];
#pragma unroll
      for (int e = 0; e < 8; ++e) s0[e * LDSP + 8]  = rb1[e];
#pragma unroll
      for (int e = 0; e < 8; ++e) s0[e * LDSP + 16] = rb2[e];
#pragma unroll
      for (int e = 0; e < 8; ++e) s0[e * LDSP + 24] = rb3[e];
    }
#if ASYNC_LDS
    if (!ACOL) wait_async0();
#endif
    __syncthreads();

    // ---- overlap: fetch tile kb+32 into registers while WMMAs run ----
    const bool more = (kb + 32 < p.K);
    ushort8 na0, na1, nb0, nb1, nb2, nb3;
    unsigned short nac[16];
    if (more) {
      nb0 = *(const ushort8*)(Bp + (kB + 0)  * p.rsB + nB);
      nb1 = *(const ushort8*)(Bp + (kB + 8)  * p.rsB + nB);
      nb2 = *(const ushort8*)(Bp + (kB + 16) * p.rsB + nB);
      nb3 = *(const ushort8*)(Bp + (kB + 24) * p.rsB + nB);
      __builtin_prefetch(Bp + 32 * p.rsB + kB * p.rsB + nB, 0, 1);  // tile kb+64
      if (!ACOL) {
#if !ASYNC_LDS
        na0 = *(const ushort8*)(Ap + offA0);
        na1 = *(const ushort8*)(Ap + offA1);
#else
        __builtin_prefetch(Ap + offA0, 0, 1);
#endif
      } else {
#pragma unroll
        for (int r = 0; r < 16; ++r) {
          int el = tid + r * 128;
          nac[r] = Ap[(el & 63) + (el >> 6) * p.csA];
        }
      }
    }

    bf16x16 a0 = ldfrag(&As[(wm + lm) * LDSP], half);
    bf16x16 a1 = ldfrag(&As[(wm + 16 + lm) * LDSP], half);
    bf16x16 b0 = ldfrag(&Bs[(wn + lm) * LDSP], half);
    bf16x16 b1 = ldfrag(&Bs[(wn + 16 + lm) * LDSP], half);
    bf16x16 b2 = ldfrag(&Bs[(wn + 32 + lm) * LDSP], half);
    bf16x16 b3 = ldfrag(&Bs[(wn + 48 + lm) * LDSP], half);

    acc[0][0] = __builtin_amdgcn_wmma_f32_16x16x32_bf16(false, a0, false, b0, (short)0, acc[0][0], false, false);
    acc[0][1] = __builtin_amdgcn_wmma_f32_16x16x32_bf16(false, a0, false, b1, (short)0, acc[0][1], false, false);
    acc[0][2] = __builtin_amdgcn_wmma_f32_16x16x32_bf16(false, a0, false, b2, (short)0, acc[0][2], false, false);
    acc[0][3] = __builtin_amdgcn_wmma_f32_16x16x32_bf16(false, a0, false, b3, (short)0, acc[0][3], false, false);
    acc[1][0] = __builtin_amdgcn_wmma_f32_16x16x32_bf16(false, a1, false, b0, (short)0, acc[1][0], false, false);
    acc[1][1] = __builtin_amdgcn_wmma_f32_16x16x32_bf16(false, a1, false, b1, (short)0, acc[1][1], false, false);
    acc[1][2] = __builtin_amdgcn_wmma_f32_16x16x32_bf16(false, a1, false, b2, (short)0, acc[1][2], false, false);
    acc[1][3] = __builtin_amdgcn_wmma_f32_16x16x32_bf16(false, a1, false, b3, (short)0, acc[1][3], false, false);

    __syncthreads();

    if (more) {
      rb0 = nb0; rb1 = nb1; rb2 = nb2; rb3 = nb3;
      Bp += 32 * p.rsB;
      if (!ACOL) {
#if !ASYNC_LDS
        ra0 = na0; ra1 = na1;
        Ap += 32;
#endif
      } else {
#pragma unroll
        for (int r = 0; r < 16; ++r) rac[r] = nac[r];
        Ap += 32 * p.csA;
      }
    }
  }

  // ---- epilogue; D layout: elem r, lane l -> M = r + 8*(l>>4), N = l&15 ----
#pragma unroll
  for (int mi = 0; mi < 2; ++mi)
#pragma unroll
    for (int ni = 0; ni < 4; ++ni)
#pragma unroll
      for (int r = 0; r < 8; ++r) {
        int i = i0 + wm + mi * 16 + r + 8 * half;
        int j = j0 + wn + ni * 16 + lm;
        float v = acc[mi][ni][r];
        if (EPI == EPI_H) {
          ((unsigned short*)p.out)[(long long)z * p.batchO + (long long)i * p.ldO + j] =
              f2bfu(v + p.bias[i]);
        } else if (EPI == EPI_F32) {
          ((float*)p.out)[(long long)z * p.batchO + (long long)i * p.ldO + j] = v;
        } else if (EPI == EPI_G) {
          ((unsigned short*)p.out)[(long long)z * p.batchO + (long long)i * p.ldO + j] = f2bfu(v);
        } else {  // EPI_OUT: out = (acc+gcn_b)*emb + skip, layout [b,c,N_,T_]
          if (j < N_) {
            int b = z / T_, t = z % T_;
            long long idx = (((long long)(b * C_ + i)) * N_ + j) * T_ + t;
            ((float*)p.out)[idx] = (v + p.bias[i]) * p.emb[i] + p.skip[idx];
          }
        }
      }
}

// ---------------- small prep kernels ----------------
__global__ void prep_convert_k(const float* convw, const float* gcnw, const float* mem,
                               unsigned short* convwb, unsigned short* gcnwb,
                               unsigned short* memb) {
  int idx = blockIdx.x * blockDim.x + threadIdx.x;
  const int S1 = C_ * C_, S2 = C_ * 2 * C_, S3 = C_ * NP;
  if (idx < S1) {
    convwb[idx] = f2bfu(convw[idx]);
  } else if (idx < S1 + S2) {
    int i = idx - S1;
    gcnwb[i] = f2bfu(gcnw[i]);
  } else if (idx < S1 + S2 + S3) {
    int i = idx - S1 - S2, c = i / NP, m = i % NP;
    memb[i] = (m < N_) ? f2bfu(mem[c * N_ + m]) : (unsigned short)0;
  }
}

// x[b,c,N_,T_] f32 -> xp[b,t,c,NP] bf16 (zero padded)
__global__ void pack_x_k(const float* x, unsigned short* xp) {
  long long idx = (long long)blockIdx.x * blockDim.x + threadIdx.x;
  const long long TOT = (long long)B_ * T_ * C_ * NP;
  if (idx >= TOT) return;
  int n = (int)(idx % NP);
  int c = (int)((idx / NP) % C_);
  int t = (int)((idx / ((long long)NP * C_)) % T_);
  int b = (int)(idx / ((long long)NP * C_ * T_));
  float v = (n < N_) ? x[(((long long)(b * C_ + c)) * N_ + n) * T_ + t] : 0.0f;
  xp[idx] = f2bfu(v);
}

// xs[b,c,n] = sum_t h[b,t,c,n]   (h bf16 [b,t,c,NP])
__global__ void reduce_xs_k(const unsigned short* h, unsigned short* xs) {
  long long idx = (long long)blockIdx.x * blockDim.x + threadIdx.x;
  const long long TOT = (long long)B_ * C_ * NP;
  if (idx >= TOT) return;
  int n = (int)(idx % NP);
  int c = (int)((idx / NP) % C_);
  int b = (int)(idx / ((long long)NP * C_));
  float s = 0.0f;
#pragma unroll
  for (int t = 0; t < T_; ++t)
    s += bfu2f(h[(((long long)(b * T_ + t)) * C_ + c) * NP + n]);
  xs[idx] = f2bfu(s);
}

// ---------------- adjacency kernel: relu/softmax/fc/softmax/top-k ----------
__device__ __forceinline__ float blk_max(float v, float* red) {
  int t = threadIdx.x;
  red[t] = v; __syncthreads();
  for (int s = 128; s > 0; s >>= 1) {
    if (t < s) red[t] = fmaxf(red[t], red[t + s]);
    __syncthreads();
  }
  float r = red[0]; __syncthreads();
  return r;
}
__device__ __forceinline__ float blk_sum(float v, float* red) {
  int t = threadIdx.x;
  red[t] = v; __syncthreads();
  for (int s = 128; s > 0; s >>= 1) {
    if (t < s) red[t] += red[t + s];
    __syncthreads();
  }
  float r = red[0]; __syncthreads();
  return r;
}
__device__ __forceinline__ int blk_sumi(int v, int* red) {
  int t = threadIdx.x;
  red[t] = v; __syncthreads();
  for (int s = 128; s > 0; s >>= 1) {
    if (t < s) red[t] += red[t + s];
    __syncthreads();
  }
  int r = red[0]; __syncthreads();
  return r;
}

__global__ __launch_bounds__(256) void adj_k(const float* L1, const float* L2,
                                             const float* fcw, const float* fcb,
                                             unsigned short* adj) {
  __shared__ float ebuf[NP];
  __shared__ float redf[256];
  __shared__ int   redi[256];
  const int n = blockIdx.x, b = blockIdx.y, tid = threadIdx.x;
  unsigned short* row = adj + ((long long)b * NP + n) * NP;
  if (n >= N_) {  // zero pad rows
    for (int m = tid; m < NP; m += 256) row[m] = 0;
    return;
  }
  const float* r1 = L1 + ((long long)b * NP + n) * NP;
  const float* r2 = L2 + ((long long)b * NP + n) * NP;

  // pass 1: maxes of relu(logits*scale)  (relu >= 0 so init 0 is exact)
  float m1 = 0.0f, m2 = 0.0f;
  for (int m = tid; m < N_; m += 256) {
    m1 = fmaxf(m1, fmaxf(r1[m] * SCALE, 0.0f));
    m2 = fmaxf(m2, fmaxf(r2[m] * SCALE, 0.0f));
  }
  float M1 = blk_max(m1, redf);
  float M2 = blk_max(m2, redf);

  // pass 2: softmax denominators
  float s1 = 0.0f, s2 = 0.0f;
  for (int m = tid; m < N_; m += 256) {
    s1 += __expf(fmaxf(r1[m] * SCALE, 0.0f) - M1);
    s2 += __expf(fmaxf(r2[m] * SCALE, 0.0f) - M2);
  }
  float S1 = blk_sum(s1, redf);
  float S2 = blk_sum(s2, redf);

  // pass 3: f = w0*a1 + w1*a2 + fb ; track max
  const float w0 = fcw[0], w1 = fcw[1], fb = fcb[0];
  float mf = -1e30f;
  for (int m = tid; m < N_; m += 256) {
    float a1 = __expf(fmaxf(r1[m] * SCALE, 0.0f) - M1) / S1;
    float a2 = __expf(fmaxf(r2[m] * SCALE, 0.0f) - M2) / S2;
    float f = w0 * a1 + w1 * a2 + fb;
    ebuf[m] = f;
    mf = fmaxf(mf, f);
  }
  float Mf = blk_max(mf, redf);

  // pass 4: e = exp(f - Mf), sum
  float se = 0.0f;
  for (int m = tid; m < N_; m += 256) {
    float e = __expf(ebuf[m] - Mf);
    ebuf[m] = e;
    se += e;
  }
  float Sf = blk_sum(se, redf);

  // pass 5: exact K-th largest threshold via binary search on float bits
  // (e in (0,1], positive floats are uint-ordered)
  unsigned lo = 0u, hi = 0x3F800001u;  // just above 1.0f
  for (int it = 0; it < 32 && (hi - lo) > 1u; ++it) {
    unsigned mid = lo + ((hi - lo) >> 1);
    int c = 0;
    for (int m = tid; m < N_; m += 256)
      c += (__builtin_bit_cast(unsigned, ebuf[m]) >= mid) ? 1 : 0;
    int tot = blk_sumi(c, redi);
    if (tot >= KTOP) lo = mid; else hi = mid;
  }
  const unsigned thr = lo;

  // pass 6: write masked, normalized row (bf16), pads zeroed
  const float inv = 1.0f / Sf;
  for (int m = tid; m < NP; m += 256) {
    if (m < N_) {
      float e = ebuf[m];
      bool keep = __builtin_bit_cast(unsigned, e) >= thr;
      row[m] = keep ? f2bfu(e * inv) : (unsigned short)0;
    } else {
      row[m] = 0;
    }
  }
}

// ---------------- launcher ----------------
extern "C" void kernel_launch(void* const* d_in, const int* in_sizes, int n_in,
                              void* d_out, int out_size, void* d_ws, size_t ws_size,
                              hipStream_t stream) {
  const float* x      = (const float*)d_in[0];
  const float* conv_w = (const float*)d_in[1];
  const float* conv_b = (const float*)d_in[2];
  const float* memory = (const float*)d_in[3];
  const float* fc_w   = (const float*)d_in[4];
  const float* fc_b   = (const float*)d_in[5];
  const float* gcn_w  = (const float*)d_in[6];
  const float* gcn_b  = (const float*)d_in[7];
  const float* emb    = (const float*)d_in[8];

  // ---- workspace carve (bytes), with aliasing ----
  const size_t SZ_H  = (size_t)B_ * T_ * C_ * NP * 2;  // h bf16 [b,t,c,NP]
  const size_t SZ_XP = SZ_H;                           // packed x (later: adj)
  const size_t SZ_XS = (size_t)B_ * C_ * NP * 2;
  const size_t SZ_CW = (size_t)C_ * C_ * 2;
  const size_t SZ_GW = (size_t)C_ * 2 * C_ * 2;
  const size_t SZ_MB = (size_t)C_ * NP * 2;
  const size_t SZ_L  = (size_t)B_ * NP * NP * 4;       // f32 logits, x2

  char* ws = (char*)d_ws;
  size_t off = 0;
  auto carve = [&](size_t sz) { char* p = ws + off; off = (off + sz + 255) & ~(size_t)255; return p; };
  unsigned short* h      = (unsigned short*)carve(SZ_H);
  unsigned short* xp     = (unsigned short*)carve(SZ_XP);  // aliased by adj later
  unsigned short* xs     = (unsigned short*)carve(SZ_XS);
  unsigned short* convwb = (unsigned short*)carve(SZ_CW);
  unsigned short* gcnwb  = (unsigned short*)carve(SZ_GW);
  unsigned short* memb   = (unsigned short*)carve(SZ_MB);
  float*          L1     = (float*)carve(SZ_L);
  float*          L2     = (float*)carve(SZ_L);
  unsigned short* adjb   = xp;                  // x-pack dead after conv GEMM
  unsigned short* gbuf   = (unsigned short*)L1; // L1+L2 (205MB) >= gbuf (176MB)

  // 1. weight conversion
  {
    int tot = C_ * C_ + C_ * 2 * C_ + C_ * NP;
    prep_convert_k<<<(tot + 255) / 256, 256, 0, stream>>>(conv_w, gcn_w, memory,
                                                          convwb, gcnwb, memb);
  }
  // 2. pack x -> bf16 [b,t,c,NP]
  {
    long long tot = (long long)B_ * T_ * C_ * NP;
    pack_x_k<<<(unsigned)((tot + 255) / 256), 256, 0, stream>>>(x, xp);
  }
  // 3. h = conv_w @ x + conv_b   (per (b,t))
  {
    GemmP p{};
    p.A = convwb; p.rsA = C_; p.csA = 1; p.batchA = 0; p.divA = 1;
    p.B = xp; p.rsB = NP; p.batchB = (long long)C_ * NP; p.divB = 1;
    p.K = C_;
    p.out = h; p.ldO = NP; p.batchO = (long long)C_ * NP;
    p.bias = conv_b;
    gemm_bf16_k<EPI_H, false><<<dim3(NP / 128, C_ / 64, B_ * T_), 128, 0, stream>>>(p);
  }
  // 4. xs = sum_t h
  {
    long long tot = (long long)B_ * C_ * NP;
    reduce_xs_k<<<(unsigned)((tot + 255) / 256), 256, 0, stream>>>(h, xs);
  }
  // 5. L1 = xs^T @ mem   (per b; A column-major, csA = NP)
  {
    GemmP p{};
    p.A = xs; p.rsA = 1; p.csA = NP; p.batchA = (long long)C_ * NP; p.divA = 1;
    p.B = memb; p.rsB = NP; p.batchB = 0; p.divB = 1;
    p.K = C_;
    p.out = L1; p.ldO = NP; p.batchO = (long long)NP * NP;
    gemm_bf16_k<EPI_F32, true><<<dim3(NP / 128, NP / 64, B_), 128, 0, stream>>>(p);
  }
  // 6. L2 = xs^T @ xs   (per b)
  {
    GemmP p{};
    p.A = xs; p.rsA = 1; p.csA = NP; p.batchA = (long long)C_ * NP; p.divA = 1;
    p.B = xs; p.rsB = NP; p.batchB = (long long)C_ * NP; p.divB = 1;
    p.K = C_;
    p.out = L2; p.ldO = NP; p.batchO = (long long)NP * NP;
    gemm_bf16_k<EPI_F32, true><<<dim3(NP / 128, NP / 64, B_), 128, 0, stream>>>(p);
  }
  // 7. adjacency: softmax/fc/softmax/top-k -> bf16 (reads L1/L2, writes adjb)
  adj_k<<<dim3(NP, B_), 256, 0, stream>>>(L1, L2, fc_w, fc_b, adjb);
  // 8. g1 = h @ adj   (per (b,t); adj batched per b) -> gbuf[...,0,:,:]
  {
    GemmP p{};
    p.A = h; p.rsA = NP; p.csA = 1; p.batchA = (long long)C_ * NP; p.divA = 1;
    p.B = adjb; p.rsB = NP; p.batchB = (long long)NP * NP; p.divB = T_;
    p.K = NP;
    p.out = gbuf; p.ldO = NP; p.batchO = 2LL * C_ * NP;
    gemm_bf16_k<EPI_G, false><<<dim3(NP / 128, C_ / 64, B_ * T_), 128, 0, stream>>>(p);
  }
  // 9. g2 = g1 @ adj -> gbuf[...,1,:,:]
  {
    GemmP p{};
    p.A = gbuf; p.rsA = NP; p.csA = 1; p.batchA = 2LL * C_ * NP; p.divA = 1;
    p.B = adjb; p.rsB = NP; p.batchB = (long long)NP * NP; p.divB = T_;
    p.K = NP;
    p.out = (void*)(gbuf + (size_t)C_ * NP); p.ldO = NP; p.batchO = 2LL * C_ * NP;
    gemm_bf16_k<EPI_G, false><<<dim3(NP / 128, C_ / 64, B_ * T_), 128, 0, stream>>>(p);
  }
  // 10. out = (gcn_w @ [g1;g2] + gcn_b) * emb + x   (per (b,t), K=256)
  {
    GemmP p{};
    p.A = gcnwb; p.rsA = 2 * C_; p.csA = 1; p.batchA = 0; p.divA = 1;
    p.B = gbuf; p.rsB = NP; p.batchB = 2LL * C_ * NP; p.divB = 1;
    p.K = 2 * C_;
    p.out = d_out; p.ldO = 0; p.batchO = 0;
    p.bias = gcn_b; p.emb = emb; p.skip = x;
    gemm_bf16_k<EPI_OUT, false><<<dim3(NP / 128, C_ / 64, B_ * T_), 128, 0, stream>>>(p);
  }
}
